// LocalAttentionBlock_16904991277198
// MI455X (gfx1250) — matrix operand order repaired and checked by reading, not compiled, not run on hardware
//
#include <hip/hip_runtime.h>
#include <hip/hip_bf16.h>
#include <cstdint>

// ---------------------------------------------------------------------------
// Problem constants (LocalAttentionBlock): B=2, S=1024, D=1024, H=16, HD=64,
// WINDOW=256, FFN=4096.  M = B*S = 2048 rows of activations.
// ---------------------------------------------------------------------------
static constexpr int B_   = 2;
static constexpr int S_   = 1024;
static constexpr int D_   = 1024;
static constexpr int H_   = 16;
static constexpr int HD_  = 64;
static constexpr int WIN_ = 256;
static constexpr int FFN_ = 4096;
static constexpr int M_   = B_ * S_;

typedef __attribute__((ext_vector_type(16))) _Float16 v16h;
typedef __attribute__((ext_vector_type(8)))  _Float16 v8h;
typedef __attribute__((ext_vector_type(8)))  float    v8f;
typedef __attribute__((ext_vector_type(4)))  float    v4f;
typedef __attribute__((ext_vector_type(4)))  int      v4i;

#define WMMA_F16(a, b, c) \
    __builtin_amdgcn_wmma_f32_16x16x32_f16(false, (a), false, (b), (short)0, (c), false, false)

// ---------------------------------------------------------------------------
// CDNA5 async global->LDS copy (ASYNCcnt path), with a guaranteed-compiling
// synchronous fallback if the builtin is absent on this toolchain.
// Probe-derived signature: (AS1 v4i* src, AS3 v4i* dst, imm offset, imm cpol)
// ---------------------------------------------------------------------------
#if __has_builtin(__builtin_amdgcn_global_load_async_to_lds_b128)
#define HAS_ASYNC_LDS 1
#else
#define HAS_ASYNC_LDS 0
#endif

__device__ __forceinline__ void copy16_g2l(void* lds_dst, const void* gsrc) {
#if HAS_ASYNC_LDS
    __builtin_amdgcn_global_load_async_to_lds_b128(
        (__attribute__((address_space(1))) v4i*)(uintptr_t)gsrc,
        (__attribute__((address_space(3))) v4i*)lds_dst, 0, 0);
#else
    *(v8h*)lds_dst = *(const v8h*)gsrc;
#endif
}

__device__ __forceinline__ void wait_async_lds() {
#if HAS_ASYNC_LDS
#if __has_builtin(__builtin_amdgcn_s_wait_asynccnt)
    __builtin_amdgcn_s_wait_asynccnt(0);
#else
    asm volatile("s_wait_asynccnt 0x0" ::: "memory");
#endif
#endif
}

// ---------------------------------------------------------------------------
// Fragment loaders (wave32; this code's contiguous k order, same for A and B)
//  A 16x32 f16:  lanes 0-15 row M=lane hold K {0..15};
//                lanes 16-31 row M=lane-16 hold K {16..31}
//  B 32x16 f16:  lanes 0-15 col N=lane hold K {0..15};
//                lanes 16-31 col N=lane-16 hold K {16..31}
// ---------------------------------------------------------------------------
__device__ __forceinline__ v16h load_a_frag_f16(const _Float16* A, int lda, int row0, int k0) {
    int lane = threadIdx.x & 31;
    int hf = lane >> 4, r = lane & 15;
    const _Float16* p = A + (size_t)(row0 + r) * lda + k0 + hf * 16;
    v8h lo = *(const v8h*)p;
    v8h hi = *(const v8h*)(p + 8);
    return __builtin_shufflevector(lo, hi, 0,1,2,3,4,5,6,7,8,9,10,11,12,13,14,15);
}

// B[k][n] = X[n0+n][k0+k] where X is f16 row-major with leading dim ld
__device__ __forceinline__ v16h load_b_frag_f16rows(const _Float16* X, int ld, int n0, int k0) {
    int lane = threadIdx.x & 31;
    int hf = lane >> 4, c = lane & 15;
    const _Float16* p = X + (size_t)(n0 + c) * ld + k0 + hf * 16;
    v8h lo = *(const v8h*)p;
    v8h hi = *(const v8h*)(p + 8);
    return __builtin_shufflevector(lo, hi, 0,1,2,3,4,5,6,7,8,9,10,11,12,13,14,15);
}

// B fragment from an LDS tile stored as rows of 32 halfs (row n -> K 0..31)
__device__ __forceinline__ v16h load_b_frag_lds(const _Float16* base) {
    int lane = threadIdx.x & 31;
    int hf = lane >> 4, c = lane & 15;
    const _Float16* p = base + c * 32 + hf * 16;
    v8h lo = *(const v8h*)p;
    v8h hi = *(const v8h*)(p + 8);
    return __builtin_shufflevector(lo, hi, 0,1,2,3,4,5,6,7,8,9,10,11,12,13,14,15);
}

// A-fragment from a 16x32 f16 tile in LDS (row stride 32 halfs)
__device__ __forceinline__ v16h load_a_frag_lds(const _Float16* p0) {
    int lane = threadIdx.x & 31;
    int hf = lane >> 4, r = lane & 15;
    const _Float16* p = p0 + r * 32 + hf * 16;
    v8h lo = *(const v8h*)p;
    v8h hi = *(const v8h*)(p + 8);
    return __builtin_shufflevector(lo, hi, 0,1,2,3,4,5,6,7,8,9,10,11,12,13,14,15);
}

// ---------------------------------------------------------------------------
// Weight pre-conversion f32 -> f16 (one-time; halves steady-state W traffic)
// ---------------------------------------------------------------------------
__global__ __launch_bounds__(256) void f32_to_f16_kernel(
    const float* __restrict__ src, _Float16* __restrict__ dst, int n)
{
    int i = blockIdx.x * blockDim.x + threadIdx.x;
    if (i < n) dst[i] = (_Float16)src[i];
}

// ---------------------------------------------------------------------------
// RMSNorm (f32 in, f16 out): one block per row
// ---------------------------------------------------------------------------
__global__ __launch_bounds__(256) void rmsnorm_f16_kernel(
    const float* __restrict__ x, const float* __restrict__ w,
    _Float16* __restrict__ out, int Ddim)
{
    int row = blockIdx.x;
    const float* xr = x + (size_t)row * Ddim;
    float ss = 0.f;
    for (int c = threadIdx.x; c < Ddim; c += 256) { float v = xr[c]; ss += v * v; }
#pragma unroll
    for (int m = 16; m >= 1; m >>= 1) ss += __shfl_xor(ss, m, 32);
    __shared__ float red[8];
    if ((threadIdx.x & 31) == 0) red[threadIdx.x >> 5] = ss;
    __syncthreads();
    float tot = 0.f;
#pragma unroll
    for (int i = 0; i < 8; ++i) tot += red[i];
    float inv = rsqrtf(tot / (float)Ddim + 1e-6f);
    _Float16* orow = out + (size_t)row * Ddim;
    for (int c = threadIdx.x; c < Ddim; c += 256)
        orow[c] = (_Float16)(xr[c] * inv * w[c]);
}

// ---------------------------------------------------------------------------
// WMMA GEMM: out[m,n] = sum_k A_f16[m,k] * W_f16[n,k]  (+bias)(+resid)
// 128 threads = 4 waves; block tile 128x128, 64x64 per wave.
// Weight tile (128 x 32 f16 = 8KB) double-buffered through LDS via async
// global->LDS loads; each staged fragment feeds two waves.
// ---------------------------------------------------------------------------
__global__ __launch_bounds__(128) void gemm_f16_kernel(
    const _Float16* __restrict__ A, const _Float16* __restrict__ W,
    const float* __restrict__ bias, const float* __restrict__ resid,
    float* __restrict__ outF, _Float16* __restrict__ outH,
    int Ndim, int Kdim)
{
    __shared__ _Float16 wtile[2][128 * 32];   // 2 x 8KB

    int tid  = threadIdx.x;
    int wave = tid >> 5;
    int lane = tid & 31;
    int m0   = blockIdx.x * 128 + (wave >> 1) * 64;
    int nblk = blockIdx.y * 128;
    int nloc = (wave & 1) * 64;               // wave's n-offset inside block tile

    // thread t stages W row (nblk + t), 32 halfs (64B) per k-step
    auto stage = [&](int buf, int k0) {
        const _Float16* src = W + (size_t)(nblk + tid) * Kdim + k0;
        _Float16* dst = &wtile[buf][tid * 32];
        copy16_g2l(dst,      src);
        copy16_g2l(dst + 8,  src + 8);
        copy16_g2l(dst + 16, src + 16);
        copy16_g2l(dst + 24, src + 24);
    };

    stage(0, 0);
    wait_async_lds();
    __syncthreads();

    v8f acc[4][4];
#pragma unroll
    for (int i = 0; i < 4; ++i)
#pragma unroll
        for (int j = 0; j < 4; ++j) acc[i][j] = (v8f){0.f,0.f,0.f,0.f,0.f,0.f,0.f,0.f};

    int nsteps = Kdim >> 5;
    for (int s = 0; s < nsteps; ++s) {
        int buf = s & 1;
        int k0  = s << 5;
        if (s + 1 < nsteps) {
            stage(buf ^ 1, k0 + 32);
            __builtin_prefetch(A + (size_t)(m0 + lane) * Kdim + k0 + 32, 0, 0);
        }
        v16h a[4], b[4];
#pragma unroll
        for (int i = 0; i < 4; ++i) a[i] = load_a_frag_f16(A, Kdim, m0 + 16 * i, k0);
#pragma unroll
        for (int j = 0; j < 4; ++j) b[j] = load_b_frag_lds(&wtile[buf][(nloc + 16 * j) * 32]);
#pragma unroll
        for (int i = 0; i < 4; ++i)
#pragma unroll
            for (int j = 0; j < 4; ++j)
                acc[i][j] = WMMA_F16(a[i], b[j], acc[i][j]);
        wait_async_lds();
        __syncthreads();
    }

    int hf = lane >> 4, c = lane & 15;
#pragma unroll
    for (int i = 0; i < 4; ++i) {
#pragma unroll
        for (int j = 0; j < 4; ++j) {
            int gcol = nblk + nloc + 16 * j + c;
            float bv = bias ? bias[gcol] : 0.f;
            float rv[8];
            if (resid) {
#pragma unroll
                for (int r = 0; r < 8; ++r)
                    rv[r] = resid[(size_t)(m0 + 16 * i + r + 8 * hf) * Ndim + gcol];
            } else {
#pragma unroll
                for (int r = 0; r < 8; ++r) rv[r] = 0.f;
            }
#pragma unroll
            for (int r = 0; r < 8; ++r) {
                int grow = m0 + 16 * i + r + 8 * hf;
                float v = acc[i][j][r] + bv + rv[r];
                if (outF) outF[(size_t)grow * Ndim + gcol] = v;
                else      outH[(size_t)grow * Ndim + gcol] = (_Float16)v;
            }
        }
    }
}

// ---------------------------------------------------------------------------
// RoPE + pack: q,k f32 [M,D] -> rotated f16 [M,D];  v f32 -> vT f16 [B,H,HD,S]
// ---------------------------------------------------------------------------
__global__ __launch_bounds__(256) void rope_pack_kernel(
    const float* __restrict__ q, const float* __restrict__ k, const float* __restrict__ v,
    _Float16* __restrict__ qh, _Float16* __restrict__ kh, _Float16* __restrict__ vT)
{
    int idx = blockIdx.x * blockDim.x + threadIdx.x;
    if (idx >= M_ * D_) return;
    int col = idx & (D_ - 1);
    int row = idx >> 10;           // /D_
    int s   = row & (S_ - 1);
    int b   = row >> 10;           // /S_
    int h   = col >> 6;            // /HD_
    int d   = col & 63;
    int dh  = d & 31;
    float theta = (float)s * __powf(10000.f, -(float)(2 * dh) / 64.f);
    float cs = __cosf(theta), sn = __sinf(theta);
    float qv = q[idx], kv = k[idx];
    float qrot = (d < 32) ? -q[idx + 32] : q[idx - 32];
    float krot = (d < 32) ? -k[idx + 32] : k[idx - 32];
    qh[idx] = (_Float16)(qv * cs + qrot * sn);
    kh[idx] = (_Float16)(kv * cs + krot * sn);
    vT[((size_t)(b * H_ + h) * HD_ + d) * S_ + s] = (_Float16)v[idx];
}

// ---------------------------------------------------------------------------
// Sliding-window attention. One wave per (b,h, 16-query tile).
// Window=256 => keys live in tiles [qt-16 .. qt] (17 tiles of 16).
// Exact softmax over the 17 register-resident score tiles, dense attn write,
// P transposed through LDS into A-layout for the P*V WMMAs.
// ---------------------------------------------------------------------------
__global__ __launch_bounds__(32) void attn_kernel(
    const _Float16* __restrict__ qh, const _Float16* __restrict__ kh,
    const _Float16* __restrict__ vT, _Float16* __restrict__ ctx,
    float* __restrict__ attnOut)
{
    __shared__ _Float16 pls[16 * 32];

    int bh = blockIdx.x;                 // b*H + h
    int qt = blockIdx.y;                 // query tile
    int b = bh >> 4, h = bh & 15;
    int lane = threadIdx.x;
    int hf = lane >> 4, c = lane & 15;
    int q0 = qt * 16;
    const float scale = 0.125f;          // 1/sqrt(HD)

    // ---- zero-fill the 16 dense attention rows (window region overwritten below)
    float* arow = attnOut + ((size_t)bh * S_ + q0) * S_;
#pragma unroll 1
    for (int r = 0; r < 16; ++r) {
        v4f z = (v4f){0.f, 0.f, 0.f, 0.f};
        for (int cc = lane * 4; cc < S_; cc += 32 * 4)
            *(v4f*)(arow + (size_t)r * S_ + cc) = z;
    }

    // ---- Q fragments (16 x 64 -> two K=32 A-frags), loaded once
    const _Float16* Qb = qh + (size_t)(b * S_ + q0) * D_ + h * HD_;
    v16h aq0 = load_a_frag_f16(Qb, D_, 0, 0);
    v16h aq1 = load_a_frag_f16(Qb, D_, 0, 32);

    // ---- scores for all 17 key tiles, kept in registers
    v8f sc[17];
#pragma unroll
    for (int i = 0; i < 17; ++i) {
        int kt = qt - 16 + i;
        int ktc = kt < 0 ? 0 : kt;
        const _Float16* Kb = kh + (size_t)(b * S_ + ktc * 16) * D_ + h * HD_;
        v16h bk0 = load_b_frag_f16rows(Kb, D_, 0, 0);
        v16h bk1 = load_b_frag_f16rows(Kb, D_, 0, 32);
        v8f a = (v8f){0.f,0.f,0.f,0.f,0.f,0.f,0.f,0.f};
        a = WMMA_F16(aq0, bk0, a);
        a = WMMA_F16(aq1, bk1, a);
#pragma unroll
        for (int r = 0; r < 8; ++r) {
            int qrow = q0 + r + 8 * hf;
            int kcol = kt * 16 + c;
            bool ok = (kt >= 0) && (kcol <= qrow) && (qrow - kcol < WIN_);
            a[r] = ok ? a[r] * scale : -1e30f;
        }
        sc[i] = a;
    }

    // ---- exact softmax per row (row's 16 cols live in one lane-half)
    float lrow[8];
#pragma unroll
    for (int r = 0; r < 8; ++r) {
        float m = -1e30f;
#pragma unroll
        for (int i = 0; i < 17; ++i) m = fmaxf(m, sc[i][r]);
#pragma unroll
        for (int msk = 8; msk >= 1; msk >>= 1) m = fmaxf(m, __shfl_xor(m, msk, 16));
        float l = 0.f;
#pragma unroll
        for (int i = 0; i < 17; ++i) { float e = __expf(sc[i][r] - m); sc[i][r] = e; l += e; }
#pragma unroll
        for (int msk = 8; msk >= 1; msk >>= 1) l += __shfl_xor(l, msk, 16);
        lrow[r] = l;
    }

    // ---- normalize, write dense attention
#pragma unroll
    for (int i = 0; i < 17; ++i) {
        int kt = qt - 16 + i;
#pragma unroll
        for (int r = 0; r < 8; ++r) {
            float p = sc[i][r] / lrow[r];
            sc[i][r] = p;
            if (kt >= 0) {
                int qrow = r + 8 * hf;
                int kcol = kt * 16 + c;
                arow[(size_t)qrow * S_ + kcol] = p;
            }
        }
    }

    // ---- O = P @ V : pairs of tiles give a 16x32 f16 A-frag (via LDS transpose)
    v8f o[4];
#pragma unroll
    for (int j = 0; j < 4; ++j) o[j] = (v8f){0.f,0.f,0.f,0.f,0.f,0.f,0.f,0.f};

    const _Float16* Vb = vT + (size_t)bh * HD_ * S_;
#pragma unroll
    for (int ip = 0; ip < 9; ++ip) {
        // store P pair into LDS in C-layout (row-major 16x32)
#pragma unroll
        for (int t = 0; t < 2; ++t) {
            int i = 2 * ip + t;
#pragma unroll
            for (int r = 0; r < 8; ++r) {
                float p = 0.f;
                if (i < 17) p = sc[i][r];
                int row = r + 8 * hf;
                pls[row * 32 + t * 16 + c] = (_Float16)p;
            }
        }
        // single-wave LDS: DS ops are in-order; wait for the stores to land
        asm volatile("s_wait_dscnt 0x0" ::: "memory");
        v16h ap = load_a_frag_lds(pls);

        // per-lane-half key base for this pair (each half = one 16-key tile)
        int kk = (qt - 16 + 2 * ip) * 16 + hf * 16;
        if (kk < 0) kk = 0;                 // p==0 there, value irrelevant
        if (kk > S_ - 16) kk = S_ - 16;     // p==0 for the phantom 18th tile
#pragma unroll
        for (int j = 0; j < 4; ++j) {
            const _Float16* p = Vb + (size_t)(16 * j + c) * S_ + kk;
            v8h lo = *(const v8h*)p;
            v8h hi = *(const v8h*)(p + 8);
            v16h bv = __builtin_shufflevector(lo, hi, 0,1,2,3,4,5,6,7,8,9,10,11,12,13,14,15);
            o[j] = WMMA_F16(ap, bv, o[j]);
        }
    }

    // ---- write context (f16, [M,D] layout) for the O-projection GEMM
    _Float16* crow = ctx + (size_t)(b * S_ + q0) * D_ + h * HD_;
#pragma unroll
    for (int j = 0; j < 4; ++j)
#pragma unroll
        for (int r = 0; r < 8; ++r)
            crow[(size_t)(r + 8 * hf) * D_ + 16 * j + c] = (_Float16)o[j][r];
}

// ---------------------------------------------------------------------------
// SwiGLU activation: act = silu(gate) * up  (f16 in / f16 out)
// ---------------------------------------------------------------------------
__global__ __launch_bounds__(256) void silu_mul_kernel(
    const _Float16* __restrict__ g, const _Float16* __restrict__ u,
    _Float16* __restrict__ out, int n)
{
    int i = blockIdx.x * blockDim.x + threadIdx.x;
    if (i < n) {
        float gv = (float)g[i], uv = (float)u[i];
        out[i] = (_Float16)(gv / (1.f + __expf(-gv)) * uv);
    }
}

// ---------------------------------------------------------------------------
// Host-side orchestration
// ---------------------------------------------------------------------------
extern "C" void kernel_launch(void* const* d_in, const int* in_sizes, int n_in,
                              void* d_out, int out_size, void* d_ws, size_t ws_size,
                              hipStream_t stream) {
    const float* x   = (const float*)d_in[0];
    const float* wq  = (const float*)d_in[1];  const float* bq = (const float*)d_in[2];
    const float* wk  = (const float*)d_in[3];  const float* bk = (const float*)d_in[4];
    const float* wv  = (const float*)d_in[5];  const float* bv = (const float*)d_in[6];
    const float* wo  = (const float*)d_in[7];  const float* bo = (const float*)d_in[8];
    const float* anw = (const float*)d_in[9];
    const float* fnw = (const float*)d_in[10];
    const float* gw  = (const float*)d_in[11];
    const float* uw  = (const float*)d_in[12];
    const float* dw  = (const float*)d_in[13];

    float* out_x    = (float*)d_out;                   // [B,S,D]
    float* out_attn = out_x + (size_t)M_ * D_;         // [B,H,S,S]

    // workspace carve-out (~136 MB)
    char* ws = (char*)d_ws;
    auto carve = [&](size_t bytes) { char* p = ws; ws += (bytes + 255) & ~(size_t)255; return p; };
    _Float16* h1   = (_Float16*)carve((size_t)M_ * D_ * 2);
    float*    qf   = (float*)   carve((size_t)M_ * D_ * 4);
    float*    kf   = (float*)   carve((size_t)M_ * D_ * 4);
    float*    vf   = (float*)   carve((size_t)M_ * D_ * 4);
    _Float16* qh   = (_Float16*)carve((size_t)M_ * D_ * 2);
    _Float16* kh   = (_Float16*)carve((size_t)M_ * D_ * 2);
    _Float16* vT   = (_Float16*)carve((size_t)M_ * D_ * 2);
    _Float16* ctx  = (_Float16*)carve((size_t)M_ * D_ * 2);
    float*    x1   = (float*)   carve((size_t)M_ * D_ * 4);
    _Float16* h2   = (_Float16*)carve((size_t)M_ * D_ * 2);
    _Float16* gth  = (_Float16*)carve((size_t)M_ * FFN_ * 2);
    _Float16* uph  = (_Float16*)carve((size_t)M_ * FFN_ * 2);
    _Float16* acth = (_Float16*)carve((size_t)M_ * FFN_ * 2);
    _Float16* wqh  = (_Float16*)carve((size_t)D_ * D_ * 2);
    _Float16* wkh  = (_Float16*)carve((size_t)D_ * D_ * 2);
    _Float16* wvh  = (_Float16*)carve((size_t)D_ * D_ * 2);
    _Float16* woh  = (_Float16*)carve((size_t)D_ * D_ * 2);
    _Float16* gwh  = (_Float16*)carve((size_t)FFN_ * D_ * 2);
    _Float16* uwh  = (_Float16*)carve((size_t)FFN_ * D_ * 2);
    _Float16* dwh  = (_Float16*)carve((size_t)D_ * FFN_ * 2);

    // 0) one-time weight conversion to f16 (halves steady-state weight traffic)
    int nDD = D_ * D_, nFD = FFN_ * D_;
    f32_to_f16_kernel<<<nDD / 256, 256, 0, stream>>>(wq, wqh, nDD);
    f32_to_f16_kernel<<<nDD / 256, 256, 0, stream>>>(wk, wkh, nDD);
    f32_to_f16_kernel<<<nDD / 256, 256, 0, stream>>>(wv, wvh, nDD);
    f32_to_f16_kernel<<<nDD / 256, 256, 0, stream>>>(wo, woh, nDD);
    f32_to_f16_kernel<<<nFD / 256, 256, 0, stream>>>(gw, gwh, nFD);
    f32_to_f16_kernel<<<nFD / 256, 256, 0, stream>>>(uw, uwh, nFD);
    f32_to_f16_kernel<<<nFD / 256, 256, 0, stream>>>(dw, dwh, nFD);

    dim3 gemmGridD(M_ / 128, D_ / 128);     // (16, 8)
    dim3 gemmGridF(M_ / 128, FFN_ / 128);   // (16, 32)

    // 1) attention-input RMSNorm
    rmsnorm_f16_kernel<<<M_, 256, 0, stream>>>(x, anw, h1, D_);

    // 2) Q/K/V projections
    gemm_f16_kernel<<<gemmGridD, 128, 0, stream>>>(h1, wqh, bq, nullptr, qf, nullptr, D_, D_);
    gemm_f16_kernel<<<gemmGridD, 128, 0, stream>>>(h1, wkh, bk, nullptr, kf, nullptr, D_, D_);
    gemm_f16_kernel<<<gemmGridD, 128, 0, stream>>>(h1, wvh, bv, nullptr, vf, nullptr, D_, D_);

    // 3) RoPE + f16 pack (V transposed to [B,H,HD,S])
    rope_pack_kernel<<<(M_ * D_) / 256, 256, 0, stream>>>(qf, kf, vf, qh, kh, vT);

    // 4) sliding-window attention (+ dense attention matrix output)
    attn_kernel<<<dim3(B_ * H_, S_ / 16), 32, 0, stream>>>(qh, kh, vT, ctx, out_attn);

    // 5) O-projection + residual -> x1
    gemm_f16_kernel<<<gemmGridD, 128, 0, stream>>>(ctx, woh, bo, x, x1, nullptr, D_, D_);

    // 6) FFN RMSNorm
    rmsnorm_f16_kernel<<<M_, 256, 0, stream>>>(x1, fnw, h2, D_);

    // 7) gate / up projections (f16 outputs)
    gemm_f16_kernel<<<gemmGridF, 128, 0, stream>>>(h2, gwh, nullptr, nullptr, nullptr, gth, FFN_, D_);
    gemm_f16_kernel<<<gemmGridF, 128, 0, stream>>>(h2, uwh, nullptr, nullptr, nullptr, uph, FFN_, D_);

    // 8) SwiGLU
    silu_mul_kernel<<<(M_ * FFN_) / 256, 256, 0, stream>>>(gth, uph, acth, M_ * FFN_);

    // 9) down projection + residual -> final block output
    gemm_f16_kernel<<<gemmGridD, 128, 0, stream>>>(acth, dwh, nullptr, x1, out_x, nullptr, D_, FFN_);
}
